// MultiheadSelfAttention_43928925503815
// MI455X (gfx1250) — matrix operand order repaired and checked by reading, not compiled, not run on hardware
//
#include <hip/hip_runtime.h>
#include <hip/hip_bf16.h>

typedef _Float16 v16h __attribute__((ext_vector_type(16)));
typedef _Float16 v8h  __attribute__((ext_vector_type(8)));
typedef _Float16 v4h  __attribute__((ext_vector_type(4)));
typedef float    v8f  __attribute__((ext_vector_type(8)));
typedef unsigned short v8u16 __attribute__((ext_vector_type(8)));
typedef unsigned int u32x4 __attribute__((ext_vector_type(4)));
typedef int          i32x4 __attribute__((ext_vector_type(4)));
typedef int          i32x8 __attribute__((ext_vector_type(8)));
typedef int          v4i   __attribute__((vector_size(16)));   // builtin's b128 unit

#define C_IN   512
#define NSP    4096          // 64*64 spatial
#define NBATCH 8
#define M_QKV  1536
#define BM 128
#define BN 128
#define BK 32
#define KT (C_IN / BK)       // 16 k-steps

// --- CDNA5 async / TDM availability probes (fallback = proven round-1 path) ---
#if defined(__has_builtin)
#  if __has_builtin(__builtin_amdgcn_tensor_load_to_lds) && \
      __has_builtin(__builtin_amdgcn_s_wait_tensorcnt)
#    define USE_TDM_A 1
#  endif
#  if __has_builtin(__builtin_amdgcn_global_load_async_to_lds_b128) && \
      __has_builtin(__builtin_amdgcn_s_wait_asynccnt)
#    define USE_ASYNC_LDS 1
#  endif
#endif

union AFrag { v8h h[2]; v16h v; };
union BFrag { uint4 q[2]; v16h v; };

__device__ __forceinline__ unsigned ldsOffsetOf(const void* p) {
  return (unsigned)(unsigned long long)(__attribute__((address_space(3))) const char*)p;
}

// ---------------------------------------------------------------------------
// Conversion kernels (f32 -> f16 staging; weight stacking)
// ---------------------------------------------------------------------------
__global__ void convert_x_kernel(const float* __restrict__ x,
                                 _Float16* __restrict__ xh, int n4) {
  for (int i = blockIdx.x * blockDim.x + threadIdx.x; i < n4;
       i += gridDim.x * blockDim.x) {
    float4 v = ((const float4*)x)[i];
    v4h o;
    o[0] = (_Float16)v.x; o[1] = (_Float16)v.y;
    o[2] = (_Float16)v.z; o[3] = (_Float16)v.w;
    *(v4h*)(xh + (size_t)i * 4) = o;
  }
}

__global__ void convert_w_kernel(const float* __restrict__ Wq, const float* __restrict__ Wk,
                                 const float* __restrict__ Wv, const float* __restrict__ Wo,
                                 const float* __restrict__ bq, const float* __restrict__ bk,
                                 const float* __restrict__ bv,
                                 _Float16* __restrict__ Wqkvh, _Float16* __restrict__ Woh,
                                 float* __restrict__ biasqkv) {
  const int NW = M_QKV * C_IN;
  const int NO = C_IN * C_IN;
  const int total = NW + NO + M_QKV;
  for (int i = blockIdx.x * blockDim.x + threadIdx.x; i < total;
       i += gridDim.x * blockDim.x) {
    if (i < NW) {
      int which = i >> 18;
      int off = i & (NO - 1);
      const float* s = (which == 0) ? Wq : (which == 1) ? Wk : Wv;
      Wqkvh[i] = (_Float16)s[off];
    } else if (i < NW + NO) {
      Woh[i - NW] = (_Float16)Wo[i - NW];
    } else {
      int o = i - NW - NO;
      int which = o >> 9;
      int off = o & 511;
      const float* s = (which == 0) ? bq : (which == 1) ? bk : bv;
      biasqkv[o] = s[off];
    }
  }
}

// ---------------------------------------------------------------------------
// WMMA GEMM: Out[b, m, n] = sum_k W[m,k] * X[b,k,n] + bias[m]  (+ residual)
//   A tile (weights) staged by the Tensor Data Mover when available: 2D D#
//   with 32x128 tile, 2B elements, LDS padding 4 DWORDs per 16 DWORDs so rows
//   land on the 80B bank-conflict-free stride. B tile staged manually as
//   packed (k,k+1) u32 pairs so B-frags are 2x ds_load_b128.
// ---------------------------------------------------------------------------
template <bool RESID>
__global__ __launch_bounds__(256)
void gemm512_wmma(const _Float16* __restrict__ W,
                  const _Float16* __restrict__ X,
                  const float* __restrict__ bias,
                  const float* __restrict__ resid,
                  float* __restrict__ outF,
                  _Float16* __restrict__ outH,
                  size_t out_bstride) {
  __shared__ __align__(16) _Float16     As[2][BM][40];   // 20 KB (80B row stride)
  __shared__ __align__(16) unsigned int Bs[2][BN][20];   // 20 KB

  const int tid  = threadIdx.x;
  const int lane = tid & 31;
  const int wave = tid >> 5;
  const int wm   = wave >> 2;
  const int wn   = wave & 3;
  const int sel  = lane >> 4;
  const int l16  = lane & 15;

  const int n0 = blockIdx.x * BN;
  const int m0 = blockIdx.y * BM;
  const int b  = blockIdx.z;

  const _Float16* Xb = X + (size_t)b * C_IN * NSP;

  v8f acc[4][2];
#pragma unroll
  for (int i = 0; i < 4; ++i)
#pragma unroll
    for (int j = 0; j < 2; ++j)
      acc[i][j] = v8f{0.f, 0.f, 0.f, 0.f, 0.f, 0.f, 0.f, 0.f};

#if USE_TDM_A
  auto fillA = [&](int buf, int kt) {
    if (wave == 0) {  // TDM issues once per executing wave; gate to one wave
      const unsigned lds = ldsOffsetOf(&As[buf][0][0]);
      const unsigned long long ga =
          (unsigned long long)(W + (size_t)m0 * C_IN + (size_t)kt * BK);
      u32x4 g0;
      g0[0] = 1u;                                   // count=1 (valid user D#)
      g0[1] = lds;                                  // lds_addr
      g0[2] = (unsigned)ga;                         // global_addr[31:0]
      g0[3] = (unsigned)((ga >> 32) & 0x1FFFFFFu)   // global_addr[56:32]
              | (2u << 30);                         // type=2 ("image")
      i32x8 g1;
      g1[0] = (int)((1u << 16)      // data_size = 2B
                  | (1u << 20)      // pad_enable
                  | (3u << 22)      // pad_interval: 16 DWORDs (64B row)
                  | (3u << 25));    // pad_amount: 4 DWORDs (16B -> 80B stride)
      g1[1] = (int)(512u << 16);    // tensor_dim0 = 512 (K elements per row)
      g1[2] = (int)(2048u << 16);   // tensor_dim1 (OOB bound, rows)
      g1[3] = (int)((unsigned)BK << 16);   // tile_dim0 = 32
      g1[4] = BM;                   // tile_dim1 = 128 rows
      g1[5] = 512;                  // tensor_dim0_stride = 512 elements
      g1[6] = 0; g1[7] = 0;
      i32x4 z4 = {0, 0, 0, 0};
#if __clang_major__ >= 23
      i32x8 z8 = {0, 0, 0, 0, 0, 0, 0, 0};
      __builtin_amdgcn_tensor_load_to_lds(g0, g1, z4, z4, z8, 0);
#else
      __builtin_amdgcn_tensor_load_to_lds(g0, g1, z4, z4, 0);
#endif
    }
  };
#else
  auto fillA = [&](int buf, int kt) {
#pragma unroll
    for (int c = 0; c < 2; ++c) {
      int chunk = tid * 2 + c;
      int row = chunk >> 2;
      int q = chunk & 3;
      v8h v = *(const v8h*)(W + (size_t)(m0 + row) * C_IN + kt * BK + q * 8);
      *(v8h*)&As[buf][row][q * 8] = v;
    }
  };
#endif

  auto fillB = [&](int buf, int kt) {
    int kp = tid >> 4;
    int n8 = (tid & 15) * 8;
    const v8u16 r0 = *(const v8u16*)(Xb + (size_t)(kt * BK + 2 * kp) * NSP + n0 + n8);
    const v8u16 r1 = *(const v8u16*)(Xb + (size_t)(kt * BK + 2 * kp + 1) * NSP + n0 + n8);
#pragma unroll
    for (int i = 0; i < 8; ++i)
      Bs[buf][n8 + i][kp] = (unsigned int)r0[i] | ((unsigned int)r1[i] << 16);
  };

  fillA(0, 0);
  fillB(0, 0);

  for (int kt = 0; kt < KT; ++kt) {
#if USE_TDM_A
    if (wave == 0) __builtin_amdgcn_s_wait_tensorcnt(0);
#endif
    __syncthreads();
    if (kt + 1 < KT) { fillA((kt + 1) & 1, kt + 1); fillB((kt + 1) & 1, kt + 1); }
    if (kt + 2 < KT) {   // global_prefetch_b8
      __builtin_prefetch(W + (size_t)(m0 + (tid >> 1)) * C_IN + (kt + 2) * BK, 0, 0);
      __builtin_prefetch(Xb + (size_t)((kt + 2) * BK + (tid >> 3)) * NSP + n0, 0, 0);
    }
    const int buf = kt & 1;

    AFrag a[4];
    BFrag bf[2];
#pragma unroll
    for (int mf = 0; mf < 4; ++mf) {
      int row = wm * 64 + mf * 16 + l16;
      int off = sel * 8;
      a[mf].h[0] = *(const v8h*)&As[buf][row][off];
      a[mf].h[1] = *(const v8h*)&As[buf][row][off + 16];
    }
#pragma unroll
    for (int nf = 0; nf < 2; ++nf) {
      int col = wn * 32 + nf * 16 + l16;
      int kp0 = sel * 4;
      bf[nf].q[0] = *(const uint4*)&Bs[buf][col][kp0];
      bf[nf].q[1] = *(const uint4*)&Bs[buf][col][kp0 + 8];
    }
#pragma unroll
    for (int mf = 0; mf < 4; ++mf)
#pragma unroll
      for (int nf = 0; nf < 2; ++nf)
        acc[mf][nf] = __builtin_amdgcn_wmma_f32_16x16x32_f16(
            false, a[mf].v, false, bf[nf].v, (short)0, acc[mf][nf], false, false);
  }

#pragma unroll
  for (int mf = 0; mf < 4; ++mf) {
#pragma unroll
    for (int nf = 0; nf < 2; ++nf) {
      const int colg = n0 + wn * 32 + nf * 16 + l16;
#pragma unroll
      for (int r = 0; r < 8; ++r) {
        const int rowg = m0 + wm * 64 + mf * 16 + sel * 8 + r;
        float v = acc[mf][nf][r] + bias[rowg];
        const size_t oidx = (size_t)b * out_bstride + (size_t)rowg * NSP + colg;
        if (RESID)
          outF[oidx] = v + resid[oidx];
        else
          outH[oidx] = (_Float16)v;
      }
    }
  }
}

// ---------------------------------------------------------------------------
// Attention core: per (b, d) compute S[i][j] = <Q_row(i), K_row(j)>/8,
// softmax over j (8 wide), y_row(i) = gamma * sum_j attn[i][j]*V_row(j).
// Q row staged once into LDS via async global->LDS loads (ASYNCcnt) so all 8
// waves read it from LDS; K/V stream from global (whole QKV fits in 192MB L2).
// ---------------------------------------------------------------------------
__global__ __launch_bounds__(256)
void attn_apply_kernel(const _Float16* __restrict__ qkv,
                       const float* __restrict__ gamma,
                       _Float16* __restrict__ yh) {
  __shared__ __align__(16) _Float16 Qs[NSP];   // 8 KB
  __shared__ float s[8][8];
  __shared__ float attn[8][8];

  const int tid = threadIdx.x;
  const int b = blockIdx.x >> 6;
  const int d = blockIdx.x & 63;
  const _Float16* base = qkv + (size_t)b * M_QKV * NSP;

  const int lane = tid & 31;
  const int j = tid >> 5;

  const _Float16* Krow = base + (size_t)(512 + j * 64 + d) * NSP;

  for (int i = 0; i < 8; ++i) {
    const _Float16* Qrow = base + (size_t)(i * 64 + d) * NSP;
    __syncthreads();   // previous iteration done reading Qs
#if USE_ASYNC_LDS
    for (int idx = tid; idx < NSP / 8; idx += 256)
      __builtin_amdgcn_global_load_async_to_lds_b128(
          (__attribute__((address_space(1))) v4i*)(Qrow + idx * 8),
          (__attribute__((address_space(3))) v4i*)&Qs[idx * 8], 0, 0);
    __builtin_amdgcn_s_wait_asynccnt(0);
#else
    for (int idx = tid; idx < NSP / 8; idx += 256)
      *(v8h*)&Qs[idx * 8] = *(const v8h*)(Qrow + idx * 8);
#endif
    __syncthreads();

    float partial = 0.f;
    for (int c = lane; c < NSP / 8; c += 32) {
      v8h q = *(const v8h*)&Qs[c * 8];
      v8h k = *(const v8h*)(Krow + c * 8);
#pragma unroll
      for (int t = 0; t < 8; ++t) partial += (float)q[t] * (float)k[t];
    }
#pragma unroll
    for (int off = 16; off > 0; off >>= 1)
      partial += __shfl_down(partial, off, 32);
    if (lane == 0) s[i][j] = partial * 0.125f;
  }
  __syncthreads();

  if (tid < 8) {
    float m = s[tid][0];
#pragma unroll
    for (int jj = 1; jj < 8; ++jj) m = fmaxf(m, s[tid][jj]);
    float e[8], sum = 0.f;
#pragma unroll
    for (int jj = 0; jj < 8; ++jj) { e[jj] = __expf(s[tid][jj] - m); sum += e[jj]; }
    const float inv = 1.f / sum;
#pragma unroll
    for (int jj = 0; jj < 8; ++jj) attn[tid][jj] = e[jj] * inv;
  }
  __syncthreads();

  const float g = gamma[0];
  for (int c = tid; c < NSP / 8; c += 256) {
    v8h vj[8];
#pragma unroll
    for (int jj = 0; jj < 8; ++jj)
      vj[jj] = *(const v8h*)(base + (size_t)(1024 + jj * 64 + d) * NSP + c * 8);
#pragma unroll
    for (int i = 0; i < 8; ++i) {
      float f[8] = {0, 0, 0, 0, 0, 0, 0, 0};
#pragma unroll
      for (int jj = 0; jj < 8; ++jj) {
        const float w = attn[i][jj] * g;
#pragma unroll
        for (int t = 0; t < 8; ++t) f[t] += w * (float)vj[jj][t];
      }
      v8h o;
#pragma unroll
      for (int t = 0; t < 8; ++t) o[t] = (_Float16)f[t];
      *(v8h*)(yh + (size_t)((size_t)b * 512 + i * 64 + d) * NSP + c * 8) = o;
    }
  }
}

// ---------------------------------------------------------------------------
// Launch
// ---------------------------------------------------------------------------
extern "C" void kernel_launch(void* const* d_in, const int* in_sizes, int n_in,
                              void* d_out, int out_size, void* d_ws, size_t ws_size,
                              hipStream_t stream) {
  const float* x  = (const float*)d_in[0];
  const float* Wq = (const float*)d_in[1];
  const float* bq = (const float*)d_in[2];
  const float* Wk = (const float*)d_in[3];
  const float* bk = (const float*)d_in[4];
  const float* Wv = (const float*)d_in[5];
  const float* bv = (const float*)d_in[6];
  const float* Wo = (const float*)d_in[7];
  const float* bo = (const float*)d_in[8];
  const float* gm = (const float*)d_in[9];
  float* out = (float*)d_out;

  char* ws = (char*)d_ws;
  _Float16* xh      = (_Float16*)(ws + 0);
  _Float16* Wqkvh   = (_Float16*)(ws + 33554432);
  _Float16* Woh     = (_Float16*)(ws + 35127296);
  float*    biasqkv = (float*)   (ws + 35651584);
  _Float16* qkvh    = (_Float16*)(ws + 35657728);
  _Float16* yh      = (_Float16*)(ws + 136321024);

  convert_w_kernel<<<512, 256, 0, stream>>>(Wq, Wk, Wv, Wo, bq, bk, bv,
                                            Wqkvh, Woh, biasqkv);
  convert_x_kernel<<<4096, 256, 0, stream>>>(x, xh, NBATCH * C_IN * NSP / 4);

  gemm512_wmma<false><<<dim3(NSP / BN, M_QKV / BM, NBATCH), 256, 0, stream>>>(
      Wqkvh, xh, biasqkv, nullptr, nullptr, qkvh, (size_t)M_QKV * NSP);

  attn_apply_kernel<<<NBATCH * 64, 256, 0, stream>>>(qkvh, gm, yh);

  gemm512_wmma<true><<<dim3(NSP / BN, C_IN / BM, NBATCH), 256, 0, stream>>>(
      Woh, yh, bo, x, out, nullptr, (size_t)C_IN * NSP);
}
